// GCN_12008728560160
// MI455X (gfx1250) — compile-verified, matrix operand order checked by reading, not run on hardware
//
#include <hip/hip_runtime.h>

// ---------------------------------------------------------------------------
// GCN layer for MI455X (gfx1250, wave32, WMMA):
//   h   = relu(x @ W1 + b1)            [bf16 WMMA, f32 accum]
//   agg = segment_sum(h[src], dst)     [L2-resident gather + f32 atomics]
//   out = agg @ W2 + b2                [bf16 WMMA, f32 accum]
// ---------------------------------------------------------------------------

typedef __attribute__((ext_vector_type(16))) __bf16 v16bf;
typedef __attribute__((ext_vector_type(16))) float  v16f;
typedef __attribute__((ext_vector_type(8)))  float  v8f;
typedef __attribute__((ext_vector_type(4)))  float  f4;

union BFrag {
    v16bf v;
    unsigned short u[16];
    uint4 q[2];
};

union BfBits {
    __bf16 b;
    unsigned short u;
};

__device__ __forceinline__ unsigned short bf16bits(float f) {
    BfBits x;
    x.b = (__bf16)f;                     // hardware f32->bf16, RTNE
    return x.u;
}

// Per-lane K index pattern for 16-bit A/B fragments (ISA 7.12.2, 16x32 bf16):
// lanes 0-15 hold K = {0..7, 16..23}, lanes 16-31 hold K = {8..15, 24..31}.
__device__ __forceinline__ int kpat(int e, int half) {
    return (e < 8) ? (half * 8 + e) : (16 + half * 8 + (e - 8));
}

// One wave computes a 16x128 tile of Out = act(X @ W + bias).
// Block = 128 threads = 4 waves = 4 tiles. W is staged to LDS in bf16,
// pre-swizzled so each B fragment is two ds_load_b128 per lane.
template <int RELU>
__global__ __launch_bounds__(128)
void node_linear(const float* __restrict__ X, const float* __restrict__ W,
                 const float* __restrict__ bias, float* __restrict__ Out,
                 int ntiles, int nrows)
{
    __shared__ __align__(16) unsigned short sW[32][32][16]; // [kt*8+nt][lane][elem], 32 KB
    __shared__ float sBias[128];

    const int tid = threadIdx.x;

    // Stage W (f32 -> bf16) into fragment-swizzled LDS layout.
    for (int s = tid; s < 32 * 32; s += 128) {
        const int f    = s >> 5;          // fragment id = kt*8 + nt
        const int l    = s & 31;          // lane within fragment
        const int kt   = f >> 3;
        const int nt   = f & 7;
        const int half = l >> 4;
        const int n    = nt * 16 + (l & 15);
        #pragma unroll
        for (int e = 0; e < 16; ++e) {
            const int k = kt * 32 + kpat(e, half);
            sW[f][l][e] = bf16bits(W[k * 128 + n]);
        }
    }
    if (tid < 128) sBias[tid] = bias[tid];
    __syncthreads();

    const int wave = tid >> 5;
    const int lane = tid & 31;
    const int tile = blockIdx.x * 4 + wave;
    if (tile >= ntiles) return;          // whole-wave uniform: EXEC stays all-1s

    const int half = lane >> 4;
    const int mrow = lane & 15;

    int arow = tile * 16 + mrow;         // A-fragment source row (clamped)
    if (arow >= nrows) arow = nrows - 1;

    v8f acc[8];
    #pragma unroll
    for (int nt = 0; nt < 8; ++nt) {
        #pragma unroll
        for (int r = 0; r < 8; ++r) acc[nt][r] = 0.0f;
    }

    #pragma unroll
    for (int kt = 0; kt < 4; ++kt) {
        // ---- A fragment: 16x32 f32 row-major -> bf16 (matches A layout) ----
        const float* xr = X + (size_t)arow * 128 + kt * 32 + half * 8;
        const f4 f0 = *(const f4*)(xr + 0);   // K = kt*32 + half*8 + 0..3
        const f4 f1 = *(const f4*)(xr + 4);   // K = kt*32 + half*8 + 4..7
        const f4 f2 = *(const f4*)(xr + 16);  // K = kt*32 + 16 + half*8 + 0..3
        const f4 f3 = *(const f4*)(xr + 20);  // K = kt*32 + 16 + half*8 + 4..7

        v16f af;
        af[ 0] = f0.x; af[ 1] = f0.y; af[ 2] = f0.z; af[ 3] = f0.w;
        af[ 4] = f1.x; af[ 5] = f1.y; af[ 6] = f1.z; af[ 7] = f1.w;
        af[ 8] = f2.x; af[ 9] = f2.y; af[10] = f2.z; af[11] = f2.w;
        af[12] = f3.x; af[13] = f3.y; af[14] = f3.z; af[15] = f3.w;

        BFrag a;
        a.v = __builtin_convertvector(af, v16bf);  // packed v_cvt -> bf16

        // ---- 8 WMMAs across the N dimension ----
        #pragma unroll
        for (int nt = 0; nt < 8; ++nt) {
            BFrag b;
            b.q[0] = *(const uint4*)&sW[kt * 8 + nt][lane][0];
            b.q[1] = *(const uint4*)&sW[kt * 8 + nt][lane][8];
            acc[nt] = __builtin_amdgcn_wmma_f32_16x16x32_bf16(
                false, a.v, false, b.v, (short)0, acc[nt], false, false);
        }
    }

    // ---- Epilogue: bias (+ReLU) + store. C/D layout: VGPR r at lane l is
    // element (M = r + 8*(l>>4), N = l&15) of the 16x16 tile. ----
    #pragma unroll
    for (int nt = 0; nt < 8; ++nt) {
        const int col = nt * 16 + mrow;
        const float bv = sBias[col];
        #pragma unroll
        for (int r = 0; r < 8; ++r) {
            float v = acc[nt][r] + bv;
            if (RELU) v = fmaxf(v, 0.0f);
            const int grow = tile * 16 + r + 8 * half;
            if (grow < nrows) Out[(size_t)grow * 128 + col] = v;
        }
    }
}

// One wave per edge: lanes cooperatively read the full 512 B row h[src]
// (coalesced float4 per lane) and scatter-add into agg[dst] with f32 atomics.
// Both h and agg fit in the 192 MB L2, so this runs at L2 atomic rate.
__global__ __launch_bounds__(256)
void scatter_add_edges(const float* __restrict__ H, const int* __restrict__ src,
                       const int* __restrict__ dst, float* __restrict__ Agg,
                       long long total)
{
    const long long stride = (long long)gridDim.x * blockDim.x;
    for (long long idx = (long long)blockIdx.x * blockDim.x + threadIdx.x;
         idx < total; idx += stride) {
        const long long e   = idx >> 5;
        const int      part = (int)(idx & 31);
        const int s = src[e];
        const int d = dst[e];
        const f4 v = *(const f4*)(H + (size_t)s * 128 + part * 4);
        float* o = Agg + (size_t)d * 128 + part * 4;
        atomicAdd(o + 0, v.x);
        atomicAdd(o + 1, v.y);
        atomicAdd(o + 2, v.z);
        atomicAdd(o + 3, v.w);
    }
}

extern "C" void kernel_launch(void* const* d_in, const int* in_sizes, int n_in,
                              void* d_out, int out_size, void* d_ws, size_t ws_size,
                              hipStream_t stream)
{
    const float* x    = (const float*)d_in[0];
    const int*   esrc = (const int*)  d_in[1];
    const int*   edst = (const int*)  d_in[2];
    const float* W1   = (const float*)d_in[3];
    const float* b1   = (const float*)d_in[4];
    const float* W2   = (const float*)d_in[5];
    const float* b2   = (const float*)d_in[6];
    float* out = (float*)d_out;

    const int N = in_sizes[0] / 128;          // 100000
    const int E = in_sizes[1];                // 1600000
    const int ntiles = (N + 15) / 16;         // 6250 (exact)
    const size_t rows_padded = (size_t)ntiles * 16;

    float* h   = (float*)d_ws;                       // [rows_padded x 128]
    float* agg = h + rows_padded * 128;              // [rows_padded x 128]

    // agg must start at zero every call (d_ws is not re-poisoned between runs).
    hipMemsetAsync(agg, 0, rows_padded * 128 * sizeof(float), stream);

    const dim3 gemm_grid((ntiles + 3) / 4);
    const dim3 gemm_block(128);

    // Phase 1: h = relu(x @ W1 + b1)
    node_linear<1><<<gemm_grid, gemm_block, 0, stream>>>(x, W1, b1, h, ntiles, N);

    // Phase 2: agg[dst] += h[src] over all edges
    const long long total = (long long)E * 32;
    scatter_add_edges<<<8192, 256, 0, stream>>>(h, esrc, edst, agg, total);

    // Phase 3: out = agg @ W2 + b2
    node_linear<0><<<gemm_grid, gemm_block, 0, stream>>>(agg, W2, b2, out, ntiles, N);
}
